// MoEBlock_8005819040113
// MI455X (gfx1250) — matrix operand order repaired and checked by reading, not compiled, not run on hardware
//
#include <hip/hip_runtime.h>
#include <math.h>

#define D_MODEL 1024
#define D_FF    2048
#define N_EXP   8
#define N_TOK   4096   // B*S = 2*2048
#define LN_EPS  1e-5f

typedef __attribute__((ext_vector_type(16))) __bf16 v16bf;
typedef __attribute__((ext_vector_type(8)))  float  v8f;

union FragU { uint4 u[2]; v16bf v; };

__device__ __forceinline__ unsigned short f2bf(float f) {
  unsigned u = __float_as_uint(f);
  u += 0x7FFFu + ((u >> 16) & 1u);   // round-to-nearest-even
  return (unsigned short)(u >> 16);
}

// gfx1250 async global->LDS copy (16B per lane), tracked by ASYNCcnt.
__device__ __forceinline__ void async_g2l_b128(unsigned ldsOff, const void* g) {
  asm volatile("global_load_async_to_lds_b128 %0, %1, off"
               :: "v"(ldsOff), "v"(g) : "memory");
}
__device__ __forceinline__ void wait_async0() {
  asm volatile("s_wait_asynccnt 0" ::: "memory");
}

// ---------------------------------------------------------------- init ----
__global__ void k_init(int* cnt) {
  if (threadIdx.x < N_EXP) cnt[threadIdx.x] = 0;
}

// ---------------------------------------------------------------- gate ----
__global__ __launch_bounds__(256) void k_gate(
    const float* __restrict__ x, const float* __restrict__ Wg,
    const float* __restrict__ bg, float* __restrict__ tkw,
    int* __restrict__ cnt, int* __restrict__ list) {
  int t = blockIdx.x * 256 + threadIdx.x;
  if (t >= N_TOK) return;
  float acc[N_EXP];
#pragma unroll
  for (int e = 0; e < N_EXP; ++e) acc[e] = 0.f;
  const float* xr = x + (size_t)t * D_MODEL;
  const float4* wrow = (const float4*)Wg;
  for (int dd = 0; dd < D_MODEL; ++dd) {
    float  xv = xr[dd];
    float4 a  = wrow[dd * 2 + 0];
    float4 b  = wrow[dd * 2 + 1];
    acc[0] += xv * a.x; acc[1] += xv * a.y; acc[2] += xv * a.z; acc[3] += xv * a.w;
    acc[4] += xv * b.x; acc[5] += xv * b.y; acc[6] += xv * b.z; acc[7] += xv * b.w;
  }
#pragma unroll
  for (int e = 0; e < N_EXP; ++e) acc[e] += bg[e];

  int i0 = 0; float v0 = acc[0];
#pragma unroll
  for (int e = 1; e < N_EXP; ++e)
    if (acc[e] > v0) { v0 = acc[e]; i0 = e; }
  int i1 = -1; float v1 = -3.4e38f;
#pragma unroll
  for (int e = 0; e < N_EXP; ++e)
    if (e != i0 && acc[e] > v1) { v1 = acc[e]; i1 = e; }

  float e1 = __expf(v1 - v0);          // v0 is the max of the two
  float s  = 1.f + e1;
  tkw[t * 2 + 0] = 1.f / s;
  tkw[t * 2 + 1] = e1 / s;

  int p0 = atomicAdd(&cnt[i0], 1);
  list[i0 * N_TOK + p0] = t * 2 + 0;
  int p1 = atomicAdd(&cnt[i1], 1);
  list[i1 * N_TOK + p1] = t * 2 + 1;
}

// ---------------------------------------------------------- convert x -----
__global__ __launch_bounds__(256) void k_cvt(const float* __restrict__ x,
                                             unsigned short* __restrict__ xb) {
  int i = blockIdx.x * 256 + threadIdx.x;   // quad index
  float4 v = ((const float4*)x)[i];
  unsigned lo = (unsigned)f2bf(v.x) | ((unsigned)f2bf(v.y) << 16);
  unsigned hi = (unsigned)f2bf(v.z) | ((unsigned)f2bf(v.w) << 16);
  ((uint2*)xb)[i] = make_uint2(lo, hi);
}

// ------------------------------------- convert + transpose weights -------
// in: W [E][R][C] fp32 (row-major) -> out: WT [E][C][R] bf16
// grid: (C/32, R/32, E), block 256 = 32x8
__global__ __launch_bounds__(256) void k_cvtW(const float* __restrict__ W,
                                              unsigned short* __restrict__ WT,
                                              int R, int C) {
  __shared__ unsigned short t[32][33];
  const int e  = blockIdx.z;
  const int c0 = blockIdx.x * 32, r0 = blockIdx.y * 32;
  const int cl = threadIdx.x & 31, rw = threadIdx.x >> 5;
  const float* src = W + (size_t)e * R * C;
  unsigned short* dst = WT + (size_t)e * R * C;
#pragma unroll
  for (int j = 0; j < 4; ++j) {
    int r = r0 + rw + j * 8;
    t[rw + j * 8][cl] = f2bf(src[(size_t)r * C + c0 + cl]);
  }
  __syncthreads();
#pragma unroll
  for (int j = 0; j < 4; ++j) {
    int c = c0 + rw + j * 8;
    dst[(size_t)c * R + r0 + cl] = t[cl][rw + j * 8];
  }
}

// --------------------------------------------------- FFN layer 1 (GEMM) ---
// h = relu(x @ W1[e] + b1[e]).  A tile async-staged to LDS (double buffer),
// shared by all 8 waves; B fragments direct from transposed bf16 weights.
// grid: (64 Mtiles, D_FF/256 Nblocks, E).  block 256 = 8 wave32.
__global__ __launch_bounds__(256) void k_ffn1(
    const unsigned short* __restrict__ xb,
    const unsigned short* __restrict__ wt1,   // [E][D_FF][D_MODEL] bf16 (transposed)
    const float* __restrict__ b1,
    const int* __restrict__ cnt, const int* __restrict__ list,
    unsigned short* __restrict__ hbuf) {
  const int e       = blockIdx.z;
  const int nBase   = blockIdx.y * 256;
  const int n_e     = cnt[e];
  const int rowBase = blockIdx.x * 64;
  if (rowBase >= n_e) return;

  __shared__ int sPair[64];
  __shared__ __align__(32) unsigned short ldsA[2][2048];  // 2 x 4KB (64x32 bf16)

  const int tid = threadIdx.x;
  if (tid < 64) {
    int r = rowBase + tid;
    sPair[tid] = (r < n_e) ? list[e * N_TOK + r] : -1;
  }
  __syncthreads();

  const int lane  = tid & 31;
  const int wv    = tid >> 5;
  const int half  = lane >> 4;
  const int nLane = lane & 15;

  // async A staging: thread -> (row sm, 8-elem k-chunk kc); 16B lands
  // contiguously in WMMA fragment order.
  const int sm = tid >> 2;
  const int kc = tid & 3;
  int spr = sPair[sm];
  const unsigned short* sga =
      xb + (size_t)((spr < 0) ? 0 : (spr >> 1)) * D_MODEL + kc * 8;
  const unsigned ldsBase = (unsigned)(uintptr_t)&ldsA[0][0];
  const unsigned sOff = ldsBase +
      (unsigned)(((sm >> 4) * 1024) + (((sm & 15) + ((kc & 1) << 4)) * 32) +
                 ((kc >> 1) << 4));

  const unsigned short* bp[2];
#pragma unroll
  for (int b = 0; b < 2; ++b) {
    int col = nBase + (wv * 2 + b) * 16 + nLane;
    bp[b] = wt1 + ((size_t)e * D_FF + col) * D_MODEL + 8 * half;
  }

  v8f acc[4][2];
#pragma unroll
  for (int m = 0; m < 4; ++m)
#pragma unroll
    for (int b = 0; b < 2; ++b) acc[m][b] = (v8f){0,0,0,0,0,0,0,0};

  // prologue: stage k0 = 0 into buffer 0
  async_g2l_b128(sOff, sga);
  wait_async0();
  __syncthreads();

  for (int k0 = 0; k0 < D_MODEL; k0 += 32) {
    const int cur = (k0 >> 5) & 1;
    if (k0 + 32 < D_MODEL)
      async_g2l_b128(sOff + ((cur ^ 1) * 4096), sga + k0 + 32);

    FragU bb[2];
#pragma unroll
    for (int b = 0; b < 2; ++b) {
      bb[b].u[0] = *(const uint4*)(bp[b] + k0);
      bb[b].u[1] = *(const uint4*)(bp[b] + k0 + 16);
    }
    if (k0 + 32 < D_MODEL) __builtin_prefetch(bp[0] + k0 + 32, 0, 3);

    v16bf a[4];
#pragma unroll
    for (int m = 0; m < 4; ++m)
      a[m] = *(const v16bf*)(&ldsA[cur][m * 512 + lane * 16]);

#pragma unroll
    for (int m = 0; m < 4; ++m)
#pragma unroll
      for (int b = 0; b < 2; ++b)
        acc[m][b] = __builtin_amdgcn_wmma_f32_16x16x32_bf16(
            false, a[m], false, bb[b].v, (short)0, acc[m][b], false, false);

    wait_async0();
    __syncthreads();
  }

  // epilogue: +bias, ReLU, bf16 -> hbuf[pair]
#pragma unroll
  for (int m = 0; m < 4; ++m) {
#pragma unroll
    for (int b = 0; b < 2; ++b) {
      int   col  = nBase + (wv * 2 + b) * 16 + nLane;
      float bias = b1[e * D_FF + col];
#pragma unroll
      for (int j = 0; j < 8; ++j) {
        int mLoc = m * 16 + j + half * 8;
        int pr   = sPair[mLoc];
        if (pr >= 0) {
          float v = fmaxf(acc[m][b][j] + bias, 0.f);
          hbuf[(size_t)pr * D_FF + col] = f2bf(v);
        }
      }
    }
  }
}

// --------------------------------------------------- FFN layer 2 (GEMM) ---
// eo = h @ W2[e] + b2[e].  grid: (64 Mtiles, D_MODEL/256 Nblocks, E)
__global__ __launch_bounds__(256) void k_ffn2(
    const unsigned short* __restrict__ hbuf,
    const unsigned short* __restrict__ wt2,   // [E][D_MODEL][D_FF] bf16 (transposed)
    const float* __restrict__ b2,
    const int* __restrict__ cnt, const int* __restrict__ list,
    float* __restrict__ eo) {
  const int e       = blockIdx.z;
  const int nBase   = blockIdx.y * 256;
  const int n_e     = cnt[e];
  const int rowBase = blockIdx.x * 64;
  if (rowBase >= n_e) return;

  __shared__ int sPair[64];
  __shared__ __align__(32) unsigned short ldsA[2][2048];

  const int tid = threadIdx.x;
  if (tid < 64) {
    int r = rowBase + tid;
    sPair[tid] = (r < n_e) ? list[e * N_TOK + r] : -1;
  }
  __syncthreads();

  const int lane  = tid & 31;
  const int wv    = tid >> 5;
  const int half  = lane >> 4;
  const int nLane = lane & 15;

  const int sm = tid >> 2;
  const int kc = tid & 3;
  int spr = sPair[sm];
  const unsigned short* sga =
      hbuf + (size_t)((spr < 0) ? 0 : spr) * D_FF + kc * 8;
  const unsigned ldsBase = (unsigned)(uintptr_t)&ldsA[0][0];
  const unsigned sOff = ldsBase +
      (unsigned)(((sm >> 4) * 1024) + (((sm & 15) + ((kc & 1) << 4)) * 32) +
                 ((kc >> 1) << 4));

  const unsigned short* bp[2];
#pragma unroll
  for (int b = 0; b < 2; ++b) {
    int col = nBase + (wv * 2 + b) * 16 + nLane;
    bp[b] = wt2 + ((size_t)e * D_MODEL + col) * D_FF + 8 * half;
  }

  v8f acc[4][2];
#pragma unroll
  for (int m = 0; m < 4; ++m)
#pragma unroll
    for (int b = 0; b < 2; ++b) acc[m][b] = (v8f){0,0,0,0,0,0,0,0};

  async_g2l_b128(sOff, sga);
  wait_async0();
  __syncthreads();

  for (int k0 = 0; k0 < D_FF; k0 += 32) {
    const int cur = (k0 >> 5) & 1;
    if (k0 + 32 < D_FF)
      async_g2l_b128(sOff + ((cur ^ 1) * 4096), sga + k0 + 32);

    FragU bb[2];
#pragma unroll
    for (int b = 0; b < 2; ++b) {
      bb[b].u[0] = *(const uint4*)(bp[b] + k0);
      bb[b].u[1] = *(const uint4*)(bp[b] + k0 + 16);
    }
    if (k0 + 32 < D_FF) __builtin_prefetch(bp[0] + k0 + 32, 0, 3);

    v16bf a[4];
#pragma unroll
    for (int m = 0; m < 4; ++m)
      a[m] = *(const v16bf*)(&ldsA[cur][m * 512 + lane * 16]);

#pragma unroll
    for (int m = 0; m < 4; ++m)
#pragma unroll
      for (int b = 0; b < 2; ++b)
        acc[m][b] = __builtin_amdgcn_wmma_f32_16x16x32_bf16(
            false, a[m], false, bb[b].v, (short)0, acc[m][b], false, false);

    wait_async0();
    __syncthreads();
  }

#pragma unroll
  for (int m = 0; m < 4; ++m) {
#pragma unroll
    for (int b = 0; b < 2; ++b) {
      int   col  = nBase + (wv * 2 + b) * 16 + nLane;
      float bias = b2[e * D_MODEL + col];
#pragma unroll
      for (int j = 0; j < 8; ++j) {
        int mLoc = m * 16 + j + half * 8;
        int pr   = sPair[mLoc];
        if (pr >= 0)
          eo[(size_t)pr * D_MODEL + col] = acc[m][b][j] + bias;
      }
    }
  }
}

// ----------------------------------------------- combine + residual + LN --
__global__ __launch_bounds__(256) void k_combine(
    const float* __restrict__ x, const float* __restrict__ eo,
    const float* __restrict__ tkw, const float* __restrict__ gamma,
    const float* __restrict__ beta, float* __restrict__ out) {
  const int t   = blockIdx.x;
  const int tid = threadIdx.x;
  const float w0 = tkw[t * 2 + 0];
  const float w1 = tkw[t * 2 + 1];
  const float* r0 = eo + (size_t)(t * 2) * D_MODEL;
  const float* r1 = r0 + D_MODEL;
  const float* xr = x + (size_t)t * D_MODEL;

  float y[4], s = 0.f, s2 = 0.f;
#pragma unroll
  for (int j = 0; j < 4; ++j) {
    int c = j * 256 + tid;
    float v = w0 * r0[c] + w1 * r1[c] + xr[c];
    y[j] = v; s += v; s2 += v * v;
  }
  __shared__ float red[256];
  red[tid] = s; __syncthreads();
  for (int o = 128; o > 0; o >>= 1) {
    if (tid < o) red[tid] += red[tid + o];
    __syncthreads();
  }
  float mu = red[0] * (1.f / D_MODEL);
  __syncthreads();
  red[tid] = s2; __syncthreads();
  for (int o = 128; o > 0; o >>= 1) {
    if (tid < o) red[tid] += red[tid + o];
    __syncthreads();
  }
  float var = red[0] * (1.f / D_MODEL) - mu * mu;
  float inv = rsqrtf(var + LN_EPS);
#pragma unroll
  for (int j = 0; j < 4; ++j) {
    int c = j * 256 + tid;
    out[(size_t)t * D_MODEL + c] = gamma[c] * (y[j] - mu) * inv + beta[c];
  }
}

// ------------------------------------------------------------------ host --
extern "C" void kernel_launch(void* const* d_in, const int* in_sizes, int n_in,
                              void* d_out, int out_size, void* d_ws, size_t ws_size,
                              hipStream_t stream) {
  const float* x     = (const float*)d_in[0];
  const float* Wg    = (const float*)d_in[1];
  const float* bg    = (const float*)d_in[2];
  const float* W1    = (const float*)d_in[3];
  const float* b1    = (const float*)d_in[4];
  const float* W2    = (const float*)d_in[5];
  const float* b2    = (const float*)d_in[6];
  const float* gamma = (const float*)d_in[7];
  const float* beta  = (const float*)d_in[8];
  float* out = (float*)d_out;

  char* ws = (char*)d_ws;
  size_t off = 0;
  int* cnt  = (int*)(ws + off);    off += 256;                         // 8 ints
  int* list = (int*)(ws + off);    off += (size_t)N_EXP * N_TOK * 4;   // 128 KB
  float* tkw = (float*)(ws + off); off += (size_t)N_TOK * 2 * 4;       // 32 KB
  unsigned short* xb = (unsigned short*)(ws + off);
  off += (size_t)N_TOK * D_MODEL * 2;                                  // 8 MB
  unsigned short* wt1 = (unsigned short*)(ws + off);
  off += (size_t)N_EXP * D_MODEL * D_FF * 2;                           // 32 MB
  unsigned short* wt2 = (unsigned short*)(ws + off);
  off += (size_t)N_EXP * D_MODEL * D_FF * 2;                           // 32 MB
  unsigned short* hbuf = (unsigned short*)(ws + off);
  off += (size_t)N_TOK * 2 * D_FF * 2;                                 // 32 MB
  float* eo = (float*)(ws + off);                                      // 32 MB

  k_init<<<1, 32, 0, stream>>>(cnt);
  k_gate<<<N_TOK / 256, 256, 0, stream>>>(x, Wg, bg, tkw, cnt, list);
  k_cvt<<<(N_TOK * D_MODEL / 4) / 256, 256, 0, stream>>>(x, xb);
  k_cvtW<<<dim3(D_FF / 32, D_MODEL / 32, N_EXP), 256, 0, stream>>>(W1, wt1, D_MODEL, D_FF);
  k_cvtW<<<dim3(D_MODEL / 32, D_FF / 32, N_EXP), 256, 0, stream>>>(W2, wt2, D_FF, D_MODEL);
  k_ffn1<<<dim3(64, D_FF / 256, N_EXP), 256, 0, stream>>>(xb, wt1, b1, cnt, list, hbuf);
  k_ffn2<<<dim3(64, D_MODEL / 256, N_EXP), 256, 0, stream>>>(hbuf, wt2, b2, cnt, list, eo);
  k_combine<<<N_TOK, 256, 0, stream>>>(x, eo, tkw, gamma, beta, out);
}
